// MultiHeadAttention_38508676776333
// MI455X (gfx1250) — matrix operand order repaired and checked by reading, not compiled, not run on hardware
//
#include <hip/hip_runtime.h>
#include <hip/hip_bf16.h>

typedef __attribute__((ext_vector_type(16))) __bf16 v16bf;
typedef __attribute__((ext_vector_type(8)))  float  v8f;
typedef __attribute__((ext_vector_type(4)))  unsigned int u32x4;
typedef __attribute__((ext_vector_type(4)))  int i32x4;
typedef __attribute__((ext_vector_type(8)))  int i32x8;

#define B_ 4
#define T_ 2048
#define C_ 1024
#define H_ 16
#define D_ 64
#define M_ (B_*T_)   // 8192

__device__ __forceinline__ unsigned short f2bf(float f) {
  unsigned int u = __float_as_uint(f);
  unsigned int r = (u + 0x7FFFu + ((u >> 16) & 1u)) >> 16;
  return (unsigned short)r;
}

union AFrag {
  v16bf v;
  unsigned int u[8];
  unsigned short s[16];
};

// Async global->LDS copy, 16B per lane (ASYNCcnt-tracked).
__device__ __forceinline__ void async_b128(unsigned lds_addr, unsigned goff,
                                           const void* base) {
  asm volatile("global_load_async_to_lds_b128 %0, %1, %2"
               :: "v"(lds_addr), "v"(goff), "s"(base) : "memory");
}
__device__ __forceinline__ void wait_async0() {
  asm volatile("s_wait_asynccnt 0x0" ::: "memory");
}

// TDM: contiguous `bytes` (multiple of 8, tile_dim0 fits 16 bits in 8B units)
// from global -> LDS, with HW padding of 4 DWORDs after every 32 DWORDs
// (reproduces a [rows][72] layout from 64-element rows). TENSORcnt-tracked.
__device__ __forceinline__ void tdm_load_padded(unsigned lds_addr,
                                                const void* gaddr,
                                                unsigned bytes) {
  unsigned long long ga = (unsigned long long)(uintptr_t)gaddr;
  const unsigned n8 = bytes >> 3;            // 8-byte elements
  u32x4 g0;
  g0[0] = 1u;                                 // count=1, user descriptor
  g0[1] = lds_addr;                           // lds_addr (bytes)
  g0[2] = (unsigned)(ga & 0xFFFFFFFFu);       // global_addr[31:0]
  g0[3] = (unsigned)((ga >> 32) & 0x01FFFFFFu) | (2u << 30);  // [56:32] | type=2
  i32x8 g1 = {};
  // data_size=3 (8B) | pad_enable | pad_interval=4 (32 DW) | pad_amount=3 (4 DW)
  g1[0] = (3 << 16) | (1 << 20) | (4 << 22) | (3 << 25);
  g1[1] = (int)((n8 & 0xFFFFu) << 16);        // tensor_dim0[15:0]
  g1[2] = (int)((n8 >> 16) & 0xFFFFu) | (1 << 16);  // tensor_dim0[31:16], tensor_dim1=1
  g1[3] = (int)((n8 & 0xFFFFu) << 16);        // tile_dim0 = n8
  g1[4] = 0;                                  // tile_dim1/2 unused
  g1[5] = (int)n8;                            // tensor_dim0_stride
  g1[6] = 0;
  g1[7] = 0;
  i32x4 z4 = {};
  i32x8 z8 = {};
  __builtin_amdgcn_tensor_load_to_lds(g0, g1, z4, z4, z8, 0);
}

// ---------------------------------------------------------------- convert
__global__ void f32_to_bf16_kernel(const float* __restrict__ src,
                                   unsigned short* __restrict__ dst, int n) {
  int i = blockIdx.x * blockDim.x + threadIdx.x;
  int stride = gridDim.x * blockDim.x;
  for (; i < n; i += stride) dst[i] = f2bf(src[i]);
}

// ---------------------------------------------------------------- GEMM
// C = A(bf16, MxK) * B(bf16, KxN) + bias, async double-buffered LDS staging.
// MODE 0: scatter into Q/K/V (B,H,T,D) bf16, Q pre-scaled by 1/sqrt(D).
// MODE 1: plain fp32 row-major output.
template<int MODE>
__global__ __launch_bounds__(256) void gemm_bf16_kernel(
    const unsigned short* __restrict__ A,
    const unsigned short* __restrict__ Bw,
    const float* __restrict__ bias,
    unsigned short* __restrict__ qout,
    unsigned short* __restrict__ kout,
    unsigned short* __restrict__ vout,
    float* __restrict__ fout,
    int M, int N, int K)
{
  __shared__ unsigned short As[2][128][40];   // row stride 80B (16B aligned)
  __shared__ unsigned short Bs[2][32][136];   // row stride 272B (16B aligned)

  const int tid   = threadIdx.x;
  const int lane  = tid & 31;
  const int wave  = tid >> 5;
  const int waveM = wave & 3;     // 4 waves along M (32 rows each)
  const int waveN = wave >> 2;    // 2 waves along N (64 cols each)
  const int mrow  = lane & 15;
  const int kh    = lane >> 4;
  const int blockN0 = blockIdx.x * 128;
  const int blockM0 = blockIdx.y * 128;

  v8f acc[2][4] = {};

  const int arow = tid >> 1;          // 2 threads per A row, 16 bf16 each
  const int acol = (tid & 1) * 16;
  const int brow = tid >> 3;          // 8 threads per B row, 16 bf16 each
  const int bcol = (tid & 7) * 16;

  auto issue_tile = [&](int k0, int buf) {
    const unsigned ldsA = (unsigned)(uintptr_t)&As[buf][arow][acol];
    const unsigned goffA = (unsigned)(((blockM0 + arow) * K + k0 + acol) * 2);
    async_b128(ldsA,      goffA,      A);
    async_b128(ldsA + 16, goffA + 16, A);
    const unsigned ldsB = (unsigned)(uintptr_t)&Bs[buf][brow][bcol];
    const unsigned goffB = (unsigned)(((k0 + brow) * N + blockN0 + bcol) * 2);
    async_b128(ldsB,      goffB,      Bw);
    async_b128(ldsB + 16, goffB + 16, Bw);
  };

  issue_tile(0, 0);
  int buf = 0;
  for (int k0 = 0; k0 < K; k0 += 32) {
    wait_async0();
    __syncthreads();
    if (k0 + 32 < K) issue_tile(k0 + 32, buf ^ 1);

    AFrag af[2];
#pragma unroll
    for (int fm = 0; fm < 2; ++fm) {
      const int r = waveM * 32 + fm * 16 + mrow;
#pragma unroll
      for (int i = 0; i < 8; ++i) {
        const int bk = ((i < 4) ? (2 * i) : (16 + 2 * (i - 4))) + kh * 8;
        af[fm].u[i] = *(const unsigned int*)&As[buf][r][bk];
      }
    }
    AFrag bfr[4];
#pragma unroll
    for (int fn = 0; fn < 4; ++fn) {
      const int cb = waveN * 64 + fn * 16 + mrow;
#pragma unroll
      for (int i = 0; i < 8; ++i) {
        const int bk = ((i < 4) ? (2 * i) : (16 + 2 * (i - 4))) + kh * 8;
        bfr[fn].s[2 * i]     = Bs[buf][bk][cb];
        bfr[fn].s[2 * i + 1] = Bs[buf][bk + 1][cb];
      }
    }
#pragma unroll
    for (int fm = 0; fm < 2; ++fm)
#pragma unroll
      for (int fn = 0; fn < 4; ++fn)
        acc[fm][fn] = __builtin_amdgcn_wmma_f32_16x16x32_bf16(
            false, af[fm].v, false, bfr[fn].v, (short)0, acc[fm][fn],
            false, false);
    buf ^= 1;
  }

#pragma unroll
  for (int fm = 0; fm < 2; ++fm) {
#pragma unroll
    for (int fn = 0; fn < 4; ++fn) {
      const int col = blockN0 + waveN * 64 + fn * 16 + mrow;
      const float bv = bias[col];
#pragma unroll
      for (int r = 0; r < 8; ++r) {
        const int row = blockM0 + waveM * 32 + fm * 16 + r + kh * 8;
        float val = acc[fm][fn][r] + bv;
        if (MODE == 0) {
          const int chunk = col >> 10;       // /C_: 0=q 1=k 2=v
          const int cc = col & (C_ - 1);
          const int h = cc >> 6;             // /D_
          const int d = cc & (D_ - 1);
          const int b = row >> 11;           // /T_
          const int t = row & (T_ - 1);
          const size_t idx = (((size_t)(b * H_ + h) * T_) + t) * D_ + d;
          if (chunk == 0)      qout[idx] = f2bf(val * 0.125f);  // 1/sqrt(64)
          else if (chunk == 1) kout[idx] = f2bf(val);
          else                 vout[idx] = f2bf(val);
        } else {
          fout[(size_t)row * N + col] = val;
        }
      }
    }
  }
}

// ---------------------------------------------------------------- flash attention
// grid: (T/128, B*H). Each wave owns 16 q rows; K/V tiles (32x64 bf16,
// contiguous 4KB) staged via TDM with HW padding into [32][72], double-buffered.
__global__ __launch_bounds__(256) void flash_attn_kernel(
    const unsigned short* __restrict__ Q,
    const unsigned short* __restrict__ Kg,
    const unsigned short* __restrict__ Vg,
    unsigned short* __restrict__ Aout)
{
  __shared__ unsigned short Ks[2][32][72];   // row stride 144B (16B aligned)
  __shared__ unsigned short Vs[2][32][72];
  __shared__ unsigned short Ps[8][16][40];

  const int tid  = threadIdx.x;
  const int lane = tid & 31;
  const int wave = tid >> 5;
  const int mrow = lane & 15;
  const int kh   = lane >> 4;
  const int qt0  = blockIdx.x * 128;
  const int bh   = blockIdx.y;
  const size_t base = (size_t)bh * T_ * D_;
  const int qrow0 = qt0 + wave * 16;

  // Q fragments (16 rows x 64 dims = 2 A-frags), kept in registers.
  AFrag qf[2];
  {
    const unsigned int* qg =
        (const unsigned int*)(Q + base + (size_t)(qrow0 + mrow) * D_);
#pragma unroll
    for (int s = 0; s < 2; ++s)
#pragma unroll
      for (int i = 0; i < 8; ++i) {
        const int bk = s * 32 + ((i < 4) ? (2 * i) : (16 + 2 * (i - 4))) + kh * 8;
        qf[s].u[i] = qg[bk >> 1];
      }
  }

  float m_run[8], l_run[8];
  v8f o[4] = {};
#pragma unroll
  for (int r = 0; r < 8; ++r) { m_run[r] = -3.0e38f; l_run[r] = 0.0f; }

  const int kvend = qt0 + 128;          // block-uniform causal bound

  auto issue_kv = [&](int kv0, int bufi) {
    tdm_load_padded((unsigned)(uintptr_t)&Ks[bufi][0][0],
                    Kg + base + (size_t)kv0 * D_, 32 * D_ * 2);
    tdm_load_padded((unsigned)(uintptr_t)&Vs[bufi][0][0],
                    Vg + base + (size_t)kv0 * D_, 32 * D_ * 2);
  };

  if (wave == 0) issue_kv(0, 0);
  int buf = 0;
  for (int kv0 = 0; kv0 < kvend; kv0 += 32) {
    __builtin_amdgcn_s_wait_tensorcnt(0);
    __syncthreads();
    if (wave == 0 && kv0 + 32 < kvend) issue_kv(kv0 + 32, buf ^ 1);

    // S = Q * K^T  (16 x 32 scores = 2 C-frags, 2 K-steps over D=64)
    v8f s[2] = {};
#pragma unroll
    for (int ks = 0; ks < 2; ++ks) {
#pragma unroll
      for (int fn = 0; fn < 2; ++fn) {
        AFrag bfr;
#pragma unroll
        for (int i = 0; i < 8; ++i)
          bfr.u[i] = *(const unsigned int*)
              &Ks[buf][fn * 16 + mrow][ks * 32 + ((i < 4) ? (2 * i) : (16 + 2 * (i - 4))) + kh * 8];
        s[fn] = __builtin_amdgcn_wmma_f32_16x16x32_bf16(
            false, qf[ks].v, false, bfr.v, (short)0, s[fn], false, false);
      }
    }

    // causal mask
#pragma unroll
    for (int fn = 0; fn < 2; ++fn) {
      const int kvp = kv0 + fn * 16 + mrow;
#pragma unroll
      for (int r = 0; r < 8; ++r) {
        const int qr = qrow0 + r + kh * 8;
        if (kvp > qr) s[fn][r] = -3.0e38f;
      }
    }

    // online softmax (row reductions across 16-lane half via shfl_xor)
#pragma unroll
    for (int r = 0; r < 8; ++r) {
      float mx = fmaxf(s[0][r], s[1][r]);
      mx = fmaxf(mx, __shfl_xor(mx, 1, 32));
      mx = fmaxf(mx, __shfl_xor(mx, 2, 32));
      mx = fmaxf(mx, __shfl_xor(mx, 4, 32));
      mx = fmaxf(mx, __shfl_xor(mx, 8, 32));
      const float mnew = fmaxf(m_run[r], mx);
      const float scale = __expf(m_run[r] - mnew);
      const float p0 = __expf(s[0][r] - mnew);
      const float p1 = __expf(s[1][r] - mnew);
      s[0][r] = p0; s[1][r] = p1;
      float rs = p0 + p1;
      rs += __shfl_xor(rs, 1, 32);
      rs += __shfl_xor(rs, 2, 32);
      rs += __shfl_xor(rs, 4, 32);
      rs += __shfl_xor(rs, 8, 32);
      l_run[r] = l_run[r] * scale + rs;
      m_run[r] = mnew;
#pragma unroll
      for (int j = 0; j < 4; ++j) o[j][r] *= scale;
    }

    // C-frag -> A-frag relayout of P through per-wave LDS patch
#pragma unroll
    for (int fn = 0; fn < 2; ++fn)
#pragma unroll
      for (int r = 0; r < 8; ++r)
        Ps[wave][r + kh * 8][fn * 16 + mrow] = f2bf(s[fn][r]);
    __syncthreads();

    AFrag pf;
#pragma unroll
    for (int i = 0; i < 8; ++i) {
      const int bk = ((i < 4) ? (2 * i) : (16 + 2 * (i - 4))) + kh * 8;
      pf.u[i] = *(const unsigned int*)&Ps[wave][mrow][bk];
    }
    // O += P * V  (K=32 kv, N=64 dims = 4 B-frags)
#pragma unroll
    for (int fn = 0; fn < 4; ++fn) {
      AFrag vfr;
#pragma unroll
      for (int i = 0; i < 8; ++i) {
        const int kv = ((i < 4) ? (2 * i) : (16 + 2 * (i - 4))) + kh * 8;
        vfr.s[2 * i]     = Vs[buf][kv][fn * 16 + mrow];
        vfr.s[2 * i + 1] = Vs[buf][kv + 1][fn * 16 + mrow];
      }
      o[fn] = __builtin_amdgcn_wmma_f32_16x16x32_bf16(
          false, pf.v, false, vfr.v, (short)0, o[fn], false, false);
    }
    buf ^= 1;
  }

  // normalize + write (B,T,H,D) == (M, C) bf16 for projection GEMM
  const int b = bh / H_;
  const int h = bh % H_;
#pragma unroll
  for (int r = 0; r < 8; ++r) {
    const float inv = 1.0f / l_run[r];
    const int t = qrow0 + r + kh * 8;
    const size_t rowbase = ((size_t)(b * T_ + t)) * C_ + h * D_;
#pragma unroll
    for (int fn = 0; fn < 4; ++fn)
      Aout[rowbase + fn * 16 + mrow] = f2bf(o[fn][r] * inv);
  }
}

// ---------------------------------------------------------------- launch
extern "C" void kernel_launch(void* const* d_in, const int* in_sizes, int n_in,
                              void* d_out, int out_size, void* d_ws, size_t ws_size,
                              hipStream_t stream) {
  (void)in_sizes; (void)n_in; (void)out_size; (void)ws_size;

  const float* x      = (const float*)d_in[0];
  const float* w_attn = (const float*)d_in[1];
  const float* b_attn = (const float*)d_in[2];
  const float* w_proj = (const float*)d_in[3];
  const float* b_proj = (const float*)d_in[4];

  char* ws = (char*)d_ws;
  size_t off = 0;
  unsigned short* xb    = (unsigned short*)(ws + off); off += (size_t)M_ * C_ * 2;
  unsigned short* wqkv  = (unsigned short*)(ws + off); off += (size_t)C_ * 3 * C_ * 2;
  unsigned short* wproj = (unsigned short*)(ws + off); off += (size_t)C_ * C_ * 2;
  unsigned short* qb    = (unsigned short*)(ws + off); off += (size_t)M_ * C_ * 2;
  unsigned short* kb    = (unsigned short*)(ws + off); off += (size_t)M_ * C_ * 2;
  unsigned short* vb    = (unsigned short*)(ws + off); off += (size_t)M_ * C_ * 2;
  unsigned short* attnb = (unsigned short*)(ws + off); off += (size_t)M_ * C_ * 2;

  f32_to_bf16_kernel<<<dim3(2048), 256, 0, stream>>>(x, xb, M_ * C_);
  f32_to_bf16_kernel<<<dim3(2048), 256, 0, stream>>>(w_attn, wqkv, C_ * 3 * C_);
  f32_to_bf16_kernel<<<dim3(1024), 256, 0, stream>>>(w_proj, wproj, C_ * C_);

  // QKV GEMM: M=8192, N=3072, K=1024
  gemm_bf16_kernel<0><<<dim3(3 * C_ / 128, M_ / 128), 256, 0, stream>>>(
      xb, wqkv, b_attn, qb, kb, vb, nullptr, M_, 3 * C_, C_);

  // flash attention
  flash_attn_kernel<<<dim3(T_ / 128, B_ * H_), 256, 0, stream>>>(qb, kb, vb, attnb);

  // projection: M=8192, N=1024, K=1024, fp32 out
  gemm_bf16_kernel<1><<<dim3(C_ / 128, M_ / 128), 256, 0, stream>>>(
      attnb, wproj, b_proj, nullptr, nullptr, nullptr, (float*)d_out, M_, C_, C_);
}
